// RelativeAttention_52261162057934
// MI455X (gfx1250) — matrix operand order repaired and checked by reading, not compiled, not run on hardware
//
#include <hip/hip_runtime.h>
#include <hip/hip_bf16.h>

typedef __attribute__((ext_vector_type(4)))  _Float16 v4h;
typedef __attribute__((ext_vector_type(8)))  _Float16 v8h;
typedef __attribute__((ext_vector_type(16))) _Float16 v16h;
typedef __attribute__((ext_vector_type(8)))  float    v8f;

#define NUM_HEADS 4
#define HD        32
#define NTOK      49
#define NPAD      64
#define DIM       128
#define SCALE     0.17677669529663687f

// LDS strides (in elements) padded for bank-conflict-free b128 fragment loads
#define XS_STRIDE 136   // x / per-head output buffer (64 x 128 used)
#define QK_STRIDE 264   // Q|K buffer (64 x 256 used)
#define VT_STRIDE 72    // V transposed: [H][32][64 used]
#define AT_STRIDE 72    // attn probs:   [H][64][64 used]

// ---- WMMA fragment loaders -------------------------------------------------
// A (16x32 f16, M x K): lane L -> row L%16; halves K = (L/16)*8 + [0..8) and 16+(L/16)*8 + [0..8)
__device__ __forceinline__ v16h frag_a(const _Float16* base, int stride, int lane) {
    int r  = lane & 15;
    int ks = (lane >> 4) << 3;
    const _Float16* p = base + r * stride + ks;
    v8h a0 = *(const v8h*)(p);
    v8h a1 = *(const v8h*)(p + 16);
    return __builtin_shufflevector(a0, a1, 0,1,2,3,4,5,6,7,8,9,10,11,12,13,14,15);
}

// B (32x16 f16, K x N) from column-major storage (i.e. B^T row-major):
// lane L -> col L%16; halves K = (L/16)*8 + [0..8) and 16+(L/16)*8 + [0..8)
__device__ __forceinline__ v16h frag_b(const _Float16* base, int stride, int lane) {
    int c  = lane & 15;
    int kb = (lane >> 4) << 3;
    const _Float16* p = base + c * stride + kb;
    v8h b0 = *(const v8h*)(p);
    v8h b1 = *(const v8h*)(p + 16);
    return __builtin_shufflevector(b0, b1, 0,1,2,3,4,5,6,7,8,9,10,11,12,13,14,15);
}

__device__ __forceinline__ v8f wmma_f16(v16h a, v16h b, v8f c) {
    return __builtin_amdgcn_wmma_f32_16x16x32_f16(false, a, false, b, (short)0, c, false, false);
}

// ---- Prep: transpose+convert weights to f16, gather relative bias ----------
__global__ void relattn_prep_kernel(const float* __restrict__ qkv_w,
                                    const float* __restrict__ wo_w,
                                    const float* __restrict__ bias_table,
                                    const int*   __restrict__ bias_index,
                                    _Float16* __restrict__ qkv_wT,
                                    _Float16* __restrict__ wo_wT,
                                    float*    __restrict__ biasB) {
    int i = blockIdx.x * 256 + threadIdx.x;
    const int NQ = 384 * 128, NW = 128 * 128, NB = NUM_HEADS * NTOK * NTOK;
    if (i < NQ) {
        int n = i >> 7, k = i & 127;
        qkv_wT[n * 128 + k] = (_Float16)qkv_w[k * 384 + n];
    } else if (i < NQ + NW) {
        int j = i - NQ;
        int n = j >> 7, k = j & 127;
        wo_wT[n * 128 + k] = (_Float16)wo_w[k * 128 + n];
    } else if (i < NQ + NW + NB) {
        int j = i - (NQ + NW);
        int h = j / (NTOK * NTOK);
        int r = j % (NTOK * NTOK);
        int q = r / NTOK, k = r % NTOK;
        biasB[j] = bias_table[bias_index[q * NTOK + k] * NUM_HEADS + h];
    }
}

// ---- Fused windowed attention: one workgroup (8 wave32) per window ---------
__global__ __launch_bounds__(256)
void relattn_fused_kernel(const float* __restrict__ x,
                          const float* __restrict__ qkv_b,
                          const float* __restrict__ wo_b,
                          const _Float16* __restrict__ qkv_wT,
                          const _Float16* __restrict__ wo_wT,
                          const float* __restrict__ biasB,
                          float* __restrict__ out) {
    __shared__ _Float16 Xs [NPAD * XS_STRIDE];                 // x f16; later per-head attn output
    __shared__ _Float16 QKs[NPAD * QK_STRIDE];                 // Q (cols 0..127) | K (cols 128..255)
    __shared__ _Float16 Vt [NUM_HEADS * HD * VT_STRIDE];       // V transposed per head
    __shared__ _Float16 At [NUM_HEADS * NPAD * AT_STRIDE];     // softmax probs f16

    const int b    = blockIdx.x;
    const int tid  = threadIdx.x;
    const int wave = tid >> 5;
    const int lane = tid & 31;
    const int ln16 = lane & 15;
    const int hi8  = (lane >> 4) << 3;

    // Phase 0: stage x (49x128 f32 -> 64x128 f16, zero padded rows), float4 loads
    {
        const float4* xb4 = (const float4*)(x + (size_t)b * (NTOK * DIM));
        for (int j = tid; j < NPAD * (DIM / 4); j += 256) {
            int r = j >> 5, c4 = j & 31;
            float4 f = {0.f, 0.f, 0.f, 0.f};
            if (r < NTOK) f = xb4[r * (DIM / 4) + c4];
            v4h hv = {(_Float16)f.x, (_Float16)f.y, (_Float16)f.z, (_Float16)f.w};
            *(v4h*)(Xs + r * XS_STRIDE + c4 * 4) = hv;
        }
    }
    __syncthreads();

    // Phase 1: QKV = Xs(64x128) @ qkv_w(128x384) + qkv_b; Q,K row-major, V transposed
    {
        int rt = wave >> 1;            // row tile 0..3
        int c0 = (wave & 1) * 12;      // 12 of 24 col tiles
        v16h a[4];
        #pragma unroll
        for (int s = 0; s < 4; ++s)
            a[s] = frag_a(Xs + rt * 16 * XS_STRIDE + s * 32, XS_STRIDE, lane);
        for (int c = c0; c < c0 + 12; ++c) {
            v8f acc = {0.f,0.f,0.f,0.f,0.f,0.f,0.f,0.f};
            #pragma unroll
            for (int s = 0; s < 4; ++s) {
                v16h bf = frag_b(qkv_wT + c * 16 * DIM + s * 32, DIM, lane);
                acc = wmma_f16(a[s], bf, acc);
            }
            int   gcol = c * 16 + ln16;
            float bias = qkv_b[gcol];
            if (c < 16) {              // Q or K region
                #pragma unroll
                for (int v = 0; v < 8; ++v) {
                    int row = rt * 16 + v + hi8;
                    QKs[row * QK_STRIDE + gcol] = (_Float16)(acc[v] + bias);
                }
            } else {                   // V region -> store transposed
                int h   = (gcol - 256) >> 5;
                int hdc = (gcol - 256) & 31;
                #pragma unroll
                for (int v = 0; v < 8; ++v) {
                    int row = rt * 16 + v + hi8;
                    Vt[h * (HD * VT_STRIDE) + hdc * VT_STRIDE + row] = (_Float16)(acc[v] + bias);
                }
            }
        }
    }
    __syncthreads();

    // Phase 2+3: scores in registers, softmax via cross-lane shuffles (no LDS scores buffer).
    // C layout: lane holds col = kc*16 + ln16; slot v holds row = rt*16 + v + hi8, so a row's
    // 64 cols live in 4 accumulators x 16 lanes of a half-group -> shfl_xor masks 1,2,4,8.
    {
        int h  = wave >> 1;
        int q0 = (wave & 1) * 2;       // row tiles q0, q0+1
        v16h aq[2];
        #pragma unroll
        for (int i = 0; i < 2; ++i)
            aq[i] = frag_a(QKs + (q0 + i) * 16 * QK_STRIDE + h * HD, QK_STRIDE, lane);
        v8f acc[2][4];
        #pragma unroll
        for (int kc = 0; kc < 4; ++kc) {
            v16h bk = frag_b(QKs + kc * 16 * QK_STRIDE + DIM + h * HD, QK_STRIDE, lane);
            #pragma unroll
            for (int i = 0; i < 2; ++i) {
                v8f z = {0.f,0.f,0.f,0.f,0.f,0.f,0.f,0.f};
                acc[i][kc] = wmma_f16(aq[i], bk, z);
            }
        }
        const float* bh = biasB + h * (NTOK * NTOK);
        #pragma unroll
        for (int i = 0; i < 2; ++i) {
            float m[8];
            #pragma unroll
            for (int v = 0; v < 8; ++v) m[v] = -3.0e38f;
            #pragma unroll
            for (int kc = 0; kc < 4; ++kc) {
                int  col = kc * 16 + ln16;
                bool cv  = col < NTOK;
                #pragma unroll
                for (int v = 0; v < 8; ++v) {
                    int q  = (q0 + i) * 16 + v + hi8;
                    int qc = (q < NTOK) ? q : (NTOK - 1);
                    float s = cv ? (acc[i][kc][v] * SCALE + bh[qc * NTOK + col]) : -3.0e38f;
                    acc[i][kc][v] = s;
                    m[v] = fmaxf(m[v], s);
                }
            }
            #pragma unroll
            for (int v = 0; v < 8; ++v) {
                #pragma unroll
                for (int msk = 1; msk < 16; msk <<= 1)
                    m[v] = fmaxf(m[v], __shfl_xor(m[v], msk, 32));
            }
            float sum[8];
            #pragma unroll
            for (int v = 0; v < 8; ++v) sum[v] = 0.0f;
            #pragma unroll
            for (int kc = 0; kc < 4; ++kc) {
                #pragma unroll
                for (int v = 0; v < 8; ++v) {
                    float e = __expf(acc[i][kc][v] - m[v]);   // masked cols -> exp(-huge) = 0
                    acc[i][kc][v] = e;
                    sum[v] += e;
                }
            }
            #pragma unroll
            for (int v = 0; v < 8; ++v) {
                #pragma unroll
                for (int msk = 1; msk < 16; msk <<= 1)
                    sum[v] += __shfl_xor(sum[v], msk, 32);
                sum[v] = 1.0f / sum[v];
            }
            #pragma unroll
            for (int kc = 0; kc < 4; ++kc) {
                int col = kc * 16 + ln16;
                #pragma unroll
                for (int v = 0; v < 8; ++v) {
                    int q = (q0 + i) * 16 + v + hi8;
                    At[h * (NPAD * AT_STRIDE) + q * AT_STRIDE + col] =
                        (_Float16)(acc[i][kc][v] * sum[v]);
                }
            }
        }
    }
    // No barrier: each wave's attn@V A-tiles read exactly the At rows it just wrote
    // (same-wave LDS RAW is enforced by DScnt waits).

    // Phase 4: out_h = attn_h(64x64) @ V_h(64x32) -> f16 into recycled Xs (64x128)
    {
        int h   = wave >> 1;
        int rt0 = (wave & 1) * 2;
        #pragma unroll
        for (int i = 0; i < 2; ++i) {
            int rt = rt0 + i;
            v16h aa[2];
            #pragma unroll
            for (int s = 0; s < 2; ++s)
                aa[s] = frag_a(At + h * (NPAD * AT_STRIDE) + rt * 16 * AT_STRIDE + s * 32,
                               AT_STRIDE, lane);
            #pragma unroll
            for (int ct = 0; ct < 2; ++ct) {
                v8f acc = {0.f,0.f,0.f,0.f,0.f,0.f,0.f,0.f};
                #pragma unroll
                for (int s = 0; s < 2; ++s) {
                    v16h bv = frag_b(Vt + h * (HD * VT_STRIDE) + ct * 16 * VT_STRIDE + s * 32,
                                     VT_STRIDE, lane);
                    acc = wmma_f16(aa[s], bv, acc);
                }
                int col = h * HD + ct * 16 + ln16;
                #pragma unroll
                for (int v = 0; v < 8; ++v) {
                    int row = rt * 16 + v + hi8;
                    Xs[row * XS_STRIDE + col] = (_Float16)acc[v];
                }
            }
        }
    }
    __syncthreads();

    // Phase 5: final = out(64x128) @ wo_w(128x128) + wo_b -> global f32, rows < 49
    {
        int rt = wave >> 1;
        int c0 = (wave & 1) * 4;
        v16h a[4];
        #pragma unroll
        for (int s = 0; s < 4; ++s)
            a[s] = frag_a(Xs + rt * 16 * XS_STRIDE + s * 32, XS_STRIDE, lane);
        float* ob = out + (size_t)b * (NTOK * DIM);
        for (int c = c0; c < c0 + 4; ++c) {
            v8f acc = {0.f,0.f,0.f,0.f,0.f,0.f,0.f,0.f};
            #pragma unroll
            for (int s = 0; s < 4; ++s) {
                v16h bf = frag_b(wo_wT + c * 16 * DIM + s * 32, DIM, lane);
                acc = wmma_f16(a[s], bf, acc);
            }
            int   gcol = c * 16 + ln16;
            float wb   = wo_b[gcol];
            #pragma unroll
            for (int v = 0; v < 8; ++v) {
                int row = rt * 16 + v + hi8;
                if (row < NTOK) ob[row * DIM + gcol] = acc[v] + wb;
            }
        }
    }
}

extern "C" void kernel_launch(void* const* d_in, const int* in_sizes, int n_in,
                              void* d_out, int out_size, void* d_ws, size_t ws_size,
                              hipStream_t stream) {
    const float* x          = (const float*)d_in[0];
    const float* qkv_w      = (const float*)d_in[1];
    const float* qkv_b      = (const float*)d_in[2];
    const float* wo_w       = (const float*)d_in[3];
    const float* wo_b       = (const float*)d_in[4];
    const float* bias_table = (const float*)d_in[5];
    const int*   bias_index = (const int*)d_in[6];

    char* ws = (char*)d_ws;
    _Float16* qkv_wT = (_Float16*)(ws);                      // 384*128*2 = 98304 B
    _Float16* wo_wT  = (_Float16*)(ws + 98304);              // 128*128*2 = 32768 B
    float*    biasB  = (float*)(ws + 98304 + 32768);         // 4*49*49*4 = 38416 B

    int prep_total = 384 * 128 + 128 * 128 + NUM_HEADS * NTOK * NTOK;
    relattn_prep_kernel<<<(prep_total + 255) / 256, 256, 0, stream>>>(
        qkv_w, wo_w, bias_table, bias_index, qkv_wT, wo_wT, biasB);

    relattn_fused_kernel<<<4096, 256, 0, stream>>>(
        x, qkv_b, wo_b, qkv_wT, wo_wT, biasB, (float*)d_out);
}
